// DHSRNN_61254823575555
// MI455X (gfx1250) — compile-verified, matrix operand order checked by reading, not run on hardware
//
#include <hip/hip_runtime.h>
#include <math.h>

// ---------------------------------------------------------------------------
// DH-SNN forward on MI455X (gfx1250): bf16 WMMA recurrent GEMM, fp32 accum.
// Dims: B=256, T=500, IN=700, H=1024, BR=4, OUT=20. K = 700+1024 -> pad 1728.
// GEMM: wave tile 32Mx32N (2x2 wmma register blocking), 4 waves/block.
// ---------------------------------------------------------------------------

#define BATCH 256
#define TSTEPS 500
#define IN_DIM 700
#define HIDDEN 1024
#define NOUT 4096           // HIDDEN*BRANCH
#define ODIM 20
#define KREAL 1724          // IN_DIM + HIDDEN
#define LDK 1728            // KREAL padded to multiple of 32

typedef __attribute__((ext_vector_type(16))) __bf16 bf16x16;
typedef __attribute__((ext_vector_type(8)))  float  v8f;

__device__ __forceinline__ unsigned short f2bf(float f) {
    union { float f; unsigned int u; } v; v.f = f;
    unsigned int u = v.u;
    unsigned int r = u + 0x7FFFu + ((u >> 16) & 1u);   // round-to-nearest-even
    return (unsigned short)(r >> 16);
}

__device__ __forceinline__ float sigm(float x) { return 1.0f / (1.0f + expf(-x)); }

// ---------------------------------------------------------------------------
// One-time: W_dense fp32 [4096][1724] -> bf16 padded [4096][1728]
// ---------------------------------------------------------------------------
__global__ void dhsnn_convert_w(const float* __restrict__ W, unsigned short* __restrict__ Wb) {
    int i = blockIdx.x * blockDim.x + threadIdx.x;
    if (i >= NOUT * LDK) return;
    int row = i / LDK, col = i % LDK;
    Wb[i] = (col < KREAL) ? f2bf(W[row * KREAL + col]) : (unsigned short)0;
}

// ---------------------------------------------------------------------------
// One-time: sigmoids of time constants
// ---------------------------------------------------------------------------
__global__ void dhsnn_init_params(const float* __restrict__ tau_n, const float* __restrict__ tau_m,
                                  const float* __restrict__ tau_m_ro,
                                  float* __restrict__ beta, float* __restrict__ alpha,
                                  float* __restrict__ alpha_ro) {
    int i = blockIdx.x * blockDim.x + threadIdx.x;
    if (i < NOUT)   beta[i]     = sigm(tau_n[i]);     // [H][BR] flat
    if (i < HIDDEN) alpha[i]    = sigm(tau_m[i]);
    if (i < ODIM)   alpha_ro[i] = sigm(tau_m_ro[i]);
}

// ---------------------------------------------------------------------------
// One-time: zero states, zero output accumulator, pack k_in for t=0
// (x row 0 as bf16, spikes = 0, K padding = 0)
// ---------------------------------------------------------------------------
__global__ void dhsnn_init_state(float* __restrict__ din, float* __restrict__ mem,
                                 float* __restrict__ mem_ro, float* __restrict__ out,
                                 unsigned short* __restrict__ kin, const float* __restrict__ x) {
    int i = blockIdx.x * blockDim.x + threadIdx.x;      // 1<<20 threads
    din[i] = 0.0f;                                      // 256*4096
    if (i < BATCH * HIDDEN) mem[i] = 0.0f;
    if (i < BATCH * ODIM) { mem_ro[i] = 0.0f; out[i] = 0.0f; }
    if (i < BATCH * LDK) {
        int b = i / LDK, c = i % LDK;
        kin[i] = (c < IN_DIM) ? f2bf(x[(size_t)b * TSTEPS * IN_DIM + c]) : (unsigned short)0;
    }
}

// ---------------------------------------------------------------------------
// Per-step GEMM: ff[256][4096] = kin[256][1728](bf16) @ Wb[4096][1728]^T + bias
// Wave tile 32M x 32N: 2x2 wmma register blocking -> 4 wmmas per 8 b128 loads
// (2 loads/wmma vs 3 in the 16x32 version). Block = 4 waves = 64M x 64N.
// Grid (4096/64, 256/64) = (64, 4) = 256 blocks.
// Fragment layout per ISA 7.12.2 (16-bit A 16x32): lane<16 holds K {0..7,16..23},
// lane>=16 holds K {8..15,24..31}; B loaded from W rows identically (B = W^T).
// ---------------------------------------------------------------------------
__global__ void __launch_bounds__(128)
dhsnn_gemm(const unsigned short* __restrict__ Wb, const unsigned short* __restrict__ kin,
           const float* __restrict__ bias, float* __restrict__ ff) {
    const int lane = threadIdx.x & 31;
    const int wave = threadIdx.x >> 5;   // 0..3
    const int wm = wave & 1;             // 0..1
    const int wn = wave >> 1;            // 0..1
    const int m0 = blockIdx.y * 64 + wm * 32;   // rows m0..m0+31 (two 16-row tiles)
    const int n0 = blockIdx.x * 64 + wn * 32;   // cols n0..n0+31 (two 16-col tiles)
    const int half = lane >> 4;
    const int r = lane & 15;

    const unsigned short* arow0 = kin + (m0 + r) * LDK + half * 8;
    const unsigned short* arow1 = arow0 + 16 * LDK;
    const unsigned short* brow0 = Wb + (size_t)(n0 + r) * LDK + half * 8;
    const unsigned short* brow1 = brow0 + 16 * (size_t)LDK;

    v8f acc00 = {};
    v8f acc01 = {};
    v8f acc10 = {};
    v8f acc11 = {};
    for (int k = 0; k < LDK; k += 32) {
        union { uint4 u[2]; bf16x16 v; } A0, A1, B0, B1;
        A0.u[0] = *(const uint4*)(arow0 + k);
        A0.u[1] = *(const uint4*)(arow0 + k + 16);
        A1.u[0] = *(const uint4*)(arow1 + k);
        A1.u[1] = *(const uint4*)(arow1 + k + 16);
        B0.u[0] = *(const uint4*)(brow0 + k);
        B0.u[1] = *(const uint4*)(brow0 + k + 16);
        B1.u[0] = *(const uint4*)(brow1 + k);
        B1.u[1] = *(const uint4*)(brow1 + k + 16);
        acc00 = __builtin_amdgcn_wmma_f32_16x16x32_bf16(false, A0.v, false, B0.v,
                                                        (short)0, acc00, false, false);
        acc01 = __builtin_amdgcn_wmma_f32_16x16x32_bf16(false, A0.v, false, B1.v,
                                                        (short)0, acc01, false, false);
        acc10 = __builtin_amdgcn_wmma_f32_16x16x32_bf16(false, A1.v, false, B0.v,
                                                        (short)0, acc10, false, false);
        acc11 = __builtin_amdgcn_wmma_f32_16x16x32_bf16(false, A1.v, false, B1.v,
                                                        (short)0, acc11, false, false);
    }

    // C layout: vgpr j -> row tile_m + 8*half + j, col = tile_n + r
    const int col0 = n0 + r;
    const int col1 = n0 + 16 + r;
    const float bv0 = bias[col0];
    const float bv1 = bias[col1];
    #pragma unroll
    for (int j = 0; j < 8; ++j) {
        const int row0 = m0 + half * 8 + j;
        const int row1 = row0 + 16;
        ff[row0 * NOUT + col0] = acc00[j] + bv0;
        ff[row0 * NOUT + col1] = acc01[j] + bv1;
        ff[row1 * NOUT + col0] = acc10[j] + bv0;
        ff[row1 * NOUT + col1] = acc11[j] + bv1;
    }
}

// ---------------------------------------------------------------------------
// Per-step elementwise: dendrite filter, membrane, spike.
// Thread (b,h): reads ff[b][4h..4h+3], updates d_in, mem, writes spike (bf16)
// into kin's recurrent columns for the NEXT step's GEMM.
// ---------------------------------------------------------------------------
__global__ void __launch_bounds__(256)
dhsnn_update(const float4* __restrict__ ff4, float4* __restrict__ din4,
             float* __restrict__ mem, unsigned short* __restrict__ kin,
             const float4* __restrict__ beta4, const float* __restrict__ alpha) {
    int i = blockIdx.x * blockDim.x + threadIdx.x;      // 0 .. 256*1024
    int b = i >> 10, h = i & 1023;

    float4 f  = ff4[i];
    float4 be = beta4[h];
    float4 d  = din4[i];
    d.x = be.x * d.x + (1.0f - be.x) * f.x;
    d.y = be.y * d.y + (1.0f - be.y) * f.y;
    d.z = be.z * d.z + (1.0f - be.z) * f.z;
    d.w = be.w * d.w + (1.0f - be.w) * f.w;
    din4[i] = d;
    float l_in = d.x + d.y + d.z + d.w;

    float a = alpha[h];
    unsigned short* sp = kin + b * LDK + IN_DIM + h;
    float spk_old = (*sp == 0x3F80u) ? 1.0f : 0.0f;
    float m = mem[i] * a + (1.0f - a) * l_in - spk_old;   // VTH=DT=1
    mem[i] = m;
    *sp = (m - 1.0f > 0.0f) ? (unsigned short)0x3F80u : (unsigned short)0;
}

// ---------------------------------------------------------------------------
// Per-step readout: mem_ro = a_ro*mem_ro + (1-a_ro)*(spk @ W_ro^T + b_ro);
// softmax over 20; accumulate for t>10. Also packs x[:,t+1,:] into kin.
// One block per batch row.
// ---------------------------------------------------------------------------
__global__ void __launch_bounds__(256)
dhsnn_readout(unsigned short* __restrict__ kin, const float* __restrict__ W_ro,
              const float* __restrict__ b_ro, const float* __restrict__ alpha_ro,
              float* __restrict__ mem_ro, float* __restrict__ out,
              const float* __restrict__ x, int t) {
    __shared__ float s[HIDDEN];
    __shared__ float dot20[32];
    __shared__ float mr[32];
    const int b = blockIdx.x;

    const unsigned short* sp = kin + b * LDK + IN_DIM;
    for (int k = threadIdx.x; k < HIDDEN; k += 256)
        s[k] = (sp[k] == 0x3F80u) ? 1.0f : 0.0f;
    __syncthreads();

    const int wave = threadIdx.x >> 5, lane = threadIdx.x & 31;
    for (int o = wave; o < ODIM; o += 8) {
        const float* wr = W_ro + o * HIDDEN;
        float p = 0.0f;
        for (int k = lane; k < HIDDEN; k += 32) p += s[k] * wr[k];
        for (int off = 16; off; off >>= 1) p += __shfl_xor(p, off, 32);
        if (lane == 0) dot20[o] = p + b_ro[o];
    }
    __syncthreads();

    if (threadIdx.x < ODIM) {
        float ar = alpha_ro[threadIdx.x];
        float m = mem_ro[b * ODIM + threadIdx.x] * ar + (1.0f - ar) * dot20[threadIdx.x];
        mem_ro[b * ODIM + threadIdx.x] = m;
        mr[threadIdx.x] = m;
    }
    __syncthreads();

    if (threadIdx.x < ODIM && t > 10) {              // keep = (t > WARMUP)
        float mx = mr[0];
        #pragma unroll
        for (int o = 1; o < ODIM; ++o) mx = fmaxf(mx, mr[o]);
        float sum = 0.0f;
        #pragma unroll
        for (int o = 0; o < ODIM; ++o) sum += expf(mr[o] - mx);
        out[b * ODIM + threadIdx.x] += expf(mr[threadIdx.x] - mx) / sum;
    }

    // pack x for next step into kin (input columns)
    if (t + 1 < TSTEPS) {
        const float* xr = x + ((size_t)b * TSTEPS + (t + 1)) * IN_DIM;
        unsigned short* kx = kin + b * LDK;
        for (int k = threadIdx.x; k < IN_DIM; k += 256)
            kx[k] = f2bf(xr[k]);
    }
}

// ---------------------------------------------------------------------------
extern "C" void kernel_launch(void* const* d_in, const int* in_sizes, int n_in,
                              void* d_out, int out_size, void* d_ws, size_t ws_size,
                              hipStream_t stream) {
    const float* x        = (const float*)d_in[0];
    const float* W_dense  = (const float*)d_in[1];
    const float* b_dense  = (const float*)d_in[2];
    const float* tau_n    = (const float*)d_in[3];
    const float* tau_m    = (const float*)d_in[4];
    const float* W_ro     = (const float*)d_in[5];
    const float* b_ro     = (const float*)d_in[6];
    const float* tau_m_ro = (const float*)d_in[7];
    float* out = (float*)d_out;

    char* ws = (char*)d_ws;
    size_t off = 0;
    auto take = [&](size_t bytes) -> char* {
        char* p = ws + off;
        off += (bytes + 255) & ~(size_t)255;
        return p;
    };
    unsigned short* Wb     = (unsigned short*)take((size_t)NOUT * LDK * 2);   // 14.2 MB
    unsigned short* kin    = (unsigned short*)take((size_t)BATCH * LDK * 2);  // 884 KB
    float* ff              = (float*)take((size_t)BATCH * NOUT * 4);          // 4 MB
    float* din             = (float*)take((size_t)BATCH * NOUT * 4);          // 4 MB
    float* mem             = (float*)take((size_t)BATCH * HIDDEN * 4);        // 1 MB
    float* mem_ro          = (float*)take((size_t)BATCH * ODIM * 4);
    float* beta            = (float*)take((size_t)NOUT * 4);
    float* alpha           = (float*)take((size_t)HIDDEN * 4);
    float* alpha_ro        = (float*)take(32 * 4);

    // one-time setup
    {
        int n = NOUT * LDK;
        dhsnn_convert_w<<<(n + 255) / 256, 256, 0, stream>>>(W_dense, Wb);
        dhsnn_init_params<<<(NOUT + 255) / 256, 256, 0, stream>>>(tau_n, tau_m, tau_m_ro,
                                                                  beta, alpha, alpha_ro);
        dhsnn_init_state<<<(BATCH * NOUT) / 256, 256, 0, stream>>>(din, mem, mem_ro, out, kin, x);
    }

    // sequential recurrence
    for (int t = 0; t < TSTEPS; ++t) {
        dhsnn_gemm<<<dim3(NOUT / 64, BATCH / 64), 128, 0, stream>>>(Wb, kin, b_dense, ff);
        dhsnn_update<<<(BATCH * HIDDEN) / 256, 256, 0, stream>>>(
            (const float4*)ff, (float4*)din, mem, kin, (const float4*)beta, alpha);
        dhsnn_readout<<<BATCH, 256, 0, stream>>>(kin, W_ro, b_ro, alpha_ro, mem_ro, out, x, t);
    }
}
